// TIGraNet_mnist_012_15513421873216
// MI455X (gfx1250) — compile-verified, matrix operand order checked by reading, not run on hardware
//
#include <hip/hip_runtime.h>
#include <hip/hip_bf16.h>
#include <cstdint>
#include <cstddef>

// ---------------------------------------------------------------------------
// TIGraNet forward for MI455X (gfx1250, wave32, WMMA).
//
// Heavy ops are [784x784] x [784 x (512*f)] GEMMs done with
// v_wmma_f32_16x16x32_f16 (fp16 in, fp32 accumulate). All operands fit in the
// 192MB L2 -> compute-bound -> fp16 WMMA path. K padded 784 -> 800.
// Each wave computes 1 row-tile x 4 col-tiles: A fragment reused 4x, and the
// 4 WMMAs per K-step use independent accumulators (4 interleaved dep chains,
// no WMMA->WMMA RAW stalls).
// ---------------------------------------------------------------------------

#define NN   784          // nodes
#define KP   800          // padded K (multiple of 32)
#define BB   512          // batch
#define ROWT 49           // 784/16 row tiles

typedef __attribute__((ext_vector_type(16))) _Float16 v16h;
typedef __attribute__((ext_vector_type(8)))  float    v8f;

// ---------------- fp16 conversion of the (shifted) Laplacian ---------------
__global__ void cvt_lap(const float* __restrict__ L, _Float16* __restrict__ Lh) {
    int idx = blockIdx.x * blockDim.x + threadIdx.x;   // over 784*800
    if (idx >= NN * KP) return;
    int r = idx / KP, c = idx % KP;
    Lh[idx] = (c < NN) ? (_Float16)L[r * NN + c] : (_Float16)0.f;
}

// ---------------- input centering -> Xt fp16 [b][k], f_in = 1 --------------
__global__ void prep_input(const float* __restrict__ in, _Float16* __restrict__ Xt) {
    __shared__ float red[256];
    int b = blockIdx.x, t = threadIdx.x;
    float s = 0.f;
    for (int n = t; n < NN; n += 256) s += in[b * NN + n];
    red[t] = s; __syncthreads();
    for (int st = 128; st > 0; st >>= 1) { if (t < st) red[t] += red[t + st]; __syncthreads(); }
    float mean = red[0] / (float)NN;
    for (int n = t; n < KP; n += 256)
        Xt[(size_t)b * KP + n] = (n < NN) ? (_Float16)(in[b * NN + n] - mean) : (_Float16)0.f;
}

// ---------------- WMMA GEMM: Y[c][n] = sum_k A[n][k] * Xt[c][k] ------------
// A: fp16 [784 x 800] row-major. Xt: fp16 [cols x 800]. Y: f32 [cols x 800].
// One wave -> 16x64 output (1 row-tile x 4 col-tiles); K loop = 25 steps of
// 16x16x32 WMMA per col-tile with the A fragment reused across all 4.
__global__ void __launch_bounds__(128)
wmma_gemm(const _Float16* __restrict__ A, const _Float16* __restrict__ Xt,
          float* __restrict__ Y, int cols) {
    const int wavesPerBlk = blockDim.x >> 5;
    int wave = blockIdx.x * wavesPerBlk + (threadIdx.x >> 5);
    int colBlocks = cols >> 6;                    // 4 col-tiles per wave
    if (wave >= ROWT * colBlocks) return;
    int tr  = wave % ROWT;                        // row tile (node dim)
    int tcb = wave / ROWT;                        // col block (4 tiles)

    int lane = threadIdx.x & 31;
    int hl = lane & 15;                           // M / N within tile
    int hi = lane >> 4;                           // half-wave K-group selector

    const _Float16* arow = A  + (size_t)(tr * 16 + hl) * KP;
    const _Float16* brow = Xt + (size_t)(tcb * 64 + hl) * KP;

    union U16 { v16h v; unsigned u[8]; };
    v8f acc0 = {}, acc1 = {}, acc2 = {}, acc3 = {};
    for (int k0 = 0; k0 < KP; k0 += 32) {
        U16 ua, ub0, ub1, ub2, ub3;
#pragma unroll
        for (int j = 0; j < 8; ++j) {
            // 16-bit A-fragment layout: VGPR j holds K pair at
            //   base = (j<4?0:16) + hi*8 + 2*(j&3)   (pairs contiguous)
            int k = k0 + ((j < 4) ? 0 : 16) + hi * 8 + 2 * (j & 3);
            ua.u[j]  = *(const unsigned*)(arow + k);
            ub0.u[j] = *(const unsigned*)(brow + k);
            ub1.u[j] = *(const unsigned*)(brow + (size_t)16 * KP + k);
            ub2.u[j] = *(const unsigned*)(brow + (size_t)32 * KP + k);
            ub3.u[j] = *(const unsigned*)(brow + (size_t)48 * KP + k);
        }
        acc0 = __builtin_amdgcn_wmma_f32_16x16x32_f16(false, ua.v, false, ub0.v, (short)0, acc0, false, false);
        acc1 = __builtin_amdgcn_wmma_f32_16x16x32_f16(false, ua.v, false, ub1.v, (short)0, acc1, false, false);
        acc2 = __builtin_amdgcn_wmma_f32_16x16x32_f16(false, ua.v, false, ub2.v, (short)0, acc2, false, false);
        acc3 = __builtin_amdgcn_wmma_f32_16x16x32_f16(false, ua.v, false, ub3.v, (short)0, acc3, false, false);
    }
    // C/D layout: VGPR r -> node = tr*16 + r + hi*8 ; col = tile base + hl
    size_t nodeOff = (size_t)tr * 16 + hi * 8;
    float* yp0 = Y + (size_t)(tcb * 64 +  0 + hl) * KP + nodeOff;
    float* yp1 = Y + (size_t)(tcb * 64 + 16 + hl) * KP + nodeOff;
    float* yp2 = Y + (size_t)(tcb * 64 + 32 + hl) * KP + nodeOff;
    float* yp3 = Y + (size_t)(tcb * 64 + 48 + hl) * KP + nodeOff;
#pragma unroll
    for (int r = 0; r < 8; ++r) yp0[r] = acc0[r];
#pragma unroll
    for (int r = 0; r < 8; ++r) yp1[r] = acc1[r];
#pragma unroll
    for (int r = 0; r < 8; ++r) yp2[r] = acc2[r];
#pragma unroll
    for (int r = 0; r < 8; ++r) yp3[r] = acc3[r];
}

// ---------------- Z init with beta ----------------------------------------
__global__ void init_z(float* __restrict__ Z, const float* __restrict__ beta, int f_out) {
    int idx = blockIdx.x * blockDim.x + threadIdx.x;
    if (idx >= BB * NN * f_out) return;
    Z[idx] = beta[idx % f_out];
}

// ---------------- alpha accumulation, m = 0 (from fp16 Xt) ----------------
__global__ void accum_xt(float* __restrict__ Z, const _Float16* __restrict__ Xt,
                         const float* __restrict__ alpha, int f_in, int f_out, int m) {
    int idx = blockIdx.x * blockDim.x + threadIdx.x;   // over B*NN
    if (idx >= BB * NN) return;
    int b = idx / NN, n = idx % NN;
    float* zp = Z + ((size_t)b * NN + n) * f_out;
    const float* al = alpha + m * f_in * f_out;
    for (int f = 0; f < f_in; ++f) {
        float v = (float)Xt[((size_t)b * f_in + f) * KP + n];
        for (int o = 0; o < f_out; ++o) zp[o] += al[f * f_out + o] * v;
    }
}

// ------ alpha accumulation for m>=1 + re-emit fp16 Xt for next L power -----
__global__ void y_accum_emit(float* __restrict__ Z, const float* __restrict__ Y,
                             _Float16* __restrict__ Xt, const float* __restrict__ alpha,
                             int f_in, int f_out, int m) {
    int idx = blockIdx.x * blockDim.x + threadIdx.x;   // over B*KP
    if (idx >= BB * KP) return;
    int b = idx / KP, n = idx % KP;
    if (n < NN) {
        float* zp = Z + ((size_t)b * NN + n) * f_out;
        const float* al = alpha + m * f_in * f_out;
        for (int f = 0; f < f_in; ++f) {
            size_t off = ((size_t)b * f_in + f) * KP + n;
            float v = Y[off];
            Xt[off] = (_Float16)v;
            for (int o = 0; o < f_out; ++o) zp[o] += al[f * f_out + o] * v;
        }
    } else {
        for (int f = 0; f < f_in; ++f) Xt[((size_t)b * f_in + f) * KP + n] = (_Float16)0.f;
    }
}

// ---------------- Z *= mask -----------------------------------------------
__global__ void mask_mul(float* __restrict__ Z, const float* __restrict__ M, int f_out) {
    int idx = blockIdx.x * blockDim.x + threadIdx.x;   // over B*NN
    if (idx >= BB * NN) return;
    float s = M[idx];
    float* zp = Z + (size_t)idx * f_out;
    for (int o = 0; o < f_out; ++o) zp[o] *= s;
}

// ---------------- emit Z -> transposed fp16 Xt [b*f_out+o][n] -------------
__global__ void emit_z(const float* __restrict__ Z, _Float16* __restrict__ Xt, int f_out) {
    int idx = blockIdx.x * blockDim.x + threadIdx.x;   // over B*KP
    if (idx >= BB * KP) return;
    int b = idx / KP, n = idx % KP;
    for (int o = 0; o < f_out; ++o)
        Xt[((size_t)b * f_out + o) * KP + n] =
            (n < NN) ? (_Float16)Z[((size_t)b * NN + n) * f_out + o] : (_Float16)0.f;
}

// ---------------- init mask to ones ---------------------------------------
__global__ void init_ones(float* __restrict__ M) {
    int idx = blockIdx.x * blockDim.x + threadIdx.x;
    if (idx < BB * NN) M[idx] = 1.f;
}

// ---------------- dynamic pooling: per-batch top-k by max|z| --------------
__global__ void pool_topk(const float* __restrict__ Z, const float* __restrict__ Mprev,
                          float* __restrict__ Mnew, int f_out, int k) {
    __shared__ float sc[KP];
    __shared__ float redv[256];
    __shared__ int   redi[256];
    int b = blockIdx.x, t = threadIdx.x;
    for (int n = t; n < KP; n += 256) {
        float s = -1e30f;
        if (n < NN) {
            const float* zp = Z + ((size_t)b * NN + n) * f_out;
            float mx = 0.f;
            for (int o = 0; o < f_out; ++o) mx = fmaxf(mx, fabsf(zp[o]));
            s = mx * Mprev[b * NN + n];
            Mnew[b * NN + n] = 0.f;
        }
        sc[n] = s;
    }
    __syncthreads();
    for (int it = 0; it < k; ++it) {
        float bv = -1e30f; int bi = 0;
        for (int n = t; n < NN; n += 256) { float v = sc[n]; if (v > bv) { bv = v; bi = n; } }
        redv[t] = bv; redi[t] = bi;
        __syncthreads();
        for (int s = 128; s > 0; s >>= 1) {
            if (t < s && redv[t + s] > redv[t]) { redv[t] = redv[t + s]; redi[t] = redi[t + s]; }
            __syncthreads();
        }
        if (t == 0) {
            int sel = redi[0];
            sc[sel] = -1e30f;
            Mnew[b * NN + sel] = Mprev[b * NN + sel];  // new * mask_prev
        }
        __syncthreads();
    }
}

// ---------------- Y -> fp16 Xt (Chebyshev t1) ------------------------------
__global__ void cvt_y_xt(const float* __restrict__ Y, _Float16* __restrict__ Xt, int cols) {
    int idx = blockIdx.x * blockDim.x + threadIdx.x;   // over cols*KP
    if (idx >= cols * KP) return;
    int n = idx % KP;
    Xt[idx] = (n < NN) ? (_Float16)Y[idx] : (_Float16)0.f;
}

// ---------------- Chebyshev step: t_next = 2*Y - t_prev --------------------
__global__ void cheb_step(const float* __restrict__ Y, const _Float16* __restrict__ Tprev,
                          _Float16* __restrict__ Tnext, int cols) {
    int idx = blockIdx.x * blockDim.x + threadIdx.x;   // over cols*KP
    if (idx >= cols * KP) return;
    int n = idx % KP;
    Tnext[idx] = (n < NN) ? (_Float16)(2.f * Y[idx] - (float)Tprev[idx]) : (_Float16)0.f;
}

// ---------------- moments over nodes: m1, m2 into s[b,180] ----------------
__global__ void moments(const _Float16* __restrict__ Xt, float* __restrict__ S, int p) {
    int idx = blockIdx.x * blockDim.x + threadIdx.x;   // over B*9
    if (idx >= BB * 9) return;
    int b = idx / 9, f = idx % 9;
    const _Float16* row = Xt + ((size_t)b * 9 + f) * KP;
    float s1 = 0.f, s2 = 0.f;
    for (int n = 0; n < NN; ++n) { float v = (float)row[n]; s1 += v; s2 += v * v; }
    float m1 = s1 / (float)NN;
    float m2 = s2 / (float)NN - m1 * m1;
    S[b * 180 + p * 18 + f]     = m1;
    S[b * 180 + p * 18 + 9 + f] = m2;
}

// ---------------- MLP layer: out = (relu)(in @ W.T + b) -------------------
__global__ void linear(const float* __restrict__ in, const float* __restrict__ W,
                       const float* __restrict__ bias, float* __restrict__ out,
                       int IN, int OUT, int doRelu) {
    int idx = blockIdx.x * blockDim.x + threadIdx.x;   // over B*OUT
    if (idx >= BB * OUT) return;
    int b = idx / OUT, j = idx % OUT;
    float acc = bias[j];
    const float* ip = in + (size_t)b * IN;
    const float* wp = W + (size_t)j * IN;
    for (int kk = 0; kk < IN; ++kk) acc += ip[kk] * wp[kk];
    out[idx] = doRelu ? fmaxf(acc, 0.f) : acc;
}

// ===========================================================================
extern "C" void kernel_launch(void* const* d_in, const int* in_sizes, int n_in,
                              void* d_out, int out_size, void* d_ws, size_t ws_size,
                              hipStream_t stream) {
    const float* input = (const float*)d_in[0];
    const float* Lap   = (const float*)d_in[1];
    const float* Lsin  = (const float*)d_in[2];
    const float* a1 = (const float*)d_in[3];  const float* be1 = (const float*)d_in[4];
    const float* a2 = (const float*)d_in[5];  const float* be2 = (const float*)d_in[6];
    const float* a3 = (const float*)d_in[7];  const float* be3 = (const float*)d_in[8];
    const float* W1 = (const float*)d_in[9];  const float* b1 = (const float*)d_in[10];
    const float* W2 = (const float*)d_in[11]; const float* b2 = (const float*)d_in[12];
    const float* W3 = (const float*)d_in[13]; const float* b3 = (const float*)d_in[14];
    const float* W4 = (const float*)d_in[15]; const float* b4 = (const float*)d_in[16];

    // -------- workspace bump allocator (total ~58 MB) ----------------------
    char* p = (char*)d_ws;
    auto alloc = [&](size_t bytes) -> void* {
        void* r = (void*)p;
        p += (bytes + 255) & ~(size_t)255;
        return r;
    };
    const int COLS9 = BB * 9;                         // 4608
    _Float16* LH  = (_Float16*)alloc((size_t)NN * KP * 2);
    _Float16* LSH = (_Float16*)alloc((size_t)NN * KP * 2);
    _Float16* XA  = (_Float16*)alloc((size_t)COLS9 * KP * 2);
    _Float16* XB  = (_Float16*)alloc((size_t)COLS9 * KP * 2);
    _Float16* XC  = (_Float16*)alloc((size_t)COLS9 * KP * 2);
    float*    Y   = (float*)   alloc((size_t)COLS9 * KP * 4);
    float*    Z   = (float*)   alloc((size_t)BB * NN * 9 * 4);
    float*    MA  = (float*)   alloc((size_t)BB * NN * 4);
    float*    MB  = (float*)   alloc((size_t)BB * NN * 4);
    float*    S   = (float*)   alloc((size_t)BB * 180 * 4);
    float*    H1  = (float*)   alloc((size_t)BB * 100 * 4);
    float*    H2  = (float*)   alloc((size_t)BB * 80 * 4);
    float*    H3  = (float*)   alloc((size_t)BB * 60 * 4);

    const int T = 256;
    auto cdiv = [](int a, int b) { return (a + b - 1) / b; };

    // -------- setup: fp16 Laplacians, centered input, ones mask ------------
    cvt_lap<<<cdiv(NN * KP, T), T, 0, stream>>>(Lap,  LH);
    cvt_lap<<<cdiv(NN * KP, T), T, 0, stream>>>(Lsin, LSH);
    prep_input<<<BB, T, 0, stream>>>(input, XA);
    init_ones<<<cdiv(BB * NN, T), T, 0, stream>>>(MA);

    // -------- spectral conv layer ------------------------------------------
    auto conv_layer = [&](int f_in, int f_out, const float* alpha, const float* beta,
                          _Float16* XTin, _Float16* XTout, float* Mprev, float* Mnext, int k) {
        int cols = BB * f_in;
        int waves = ROWT * (cols >> 6);             // 4 col-tiles per wave
        init_z<<<cdiv(BB * NN * f_out, T), T, 0, stream>>>(Z, beta, f_out);
        accum_xt<<<cdiv(BB * NN, T), T, 0, stream>>>(Z, XTin, alpha, f_in, f_out, 0);
        for (int m = 1; m <= 4; ++m) {
            wmma_gemm<<<cdiv(waves, 4), 128, 0, stream>>>(LH, XTin, Y, cols);
            y_accum_emit<<<cdiv(BB * KP, T), T, 0, stream>>>(Z, Y, XTin, alpha, f_in, f_out, m);
        }
        mask_mul<<<cdiv(BB * NN, T), T, 0, stream>>>(Z, Mprev, f_out);
        pool_topk<<<BB, T, 0, stream>>>(Z, Mprev, Mnext, f_out, k);
        emit_z<<<cdiv(BB * KP, T), T, 0, stream>>>(Z, XTout, f_out);
    };

    conv_layer(1, 3, a1, be1, XA, XB, MA, MB, 200);   // z1 -> XB, m1 -> MB
    conv_layer(3, 6, a2, be2, XB, XA, MB, MA, 100);   // z2 -> XA, m2 -> MA
    conv_layer(6, 9, a3, be3, XA, XB, MA, MB, 50);    // z3 -> XB, m3 -> MB

    // xm = z3 * m3  (Z still holds z3; apply m3 and emit as 4608-col fp16)
    mask_mul<<<cdiv(BB * NN, T), T, 0, stream>>>(Z, MB, 9);
    emit_z<<<cdiv(BB * KP, T), T, 0, stream>>>(Z, XA, 9);

    // -------- Chebyshev statistic ------------------------------------------
    int waves9 = ROWT * (COLS9 >> 6);
    moments<<<cdiv(BB * 9, T), T, 0, stream>>>(XA, S, 0);                 // t0
    wmma_gemm<<<cdiv(waves9, 4), 128, 0, stream>>>(LSH, XA, Y, COLS9);    // t1 = Ls*t0
    cvt_y_xt<<<cdiv(COLS9 * KP, T), T, 0, stream>>>(Y, XB, COLS9);
    moments<<<cdiv(BB * 9, T), T, 0, stream>>>(XB, S, 1);

    _Float16* tp = XA; _Float16* tc = XB; _Float16* tn = XC;
    for (int pd = 2; pd <= 9; ++pd) {
        wmma_gemm<<<cdiv(waves9, 4), 128, 0, stream>>>(LSH, tc, Y, COLS9);
        cheb_step<<<cdiv(COLS9 * KP, T), T, 0, stream>>>(Y, tp, tn, COLS9);
        moments<<<cdiv(BB * 9, T), T, 0, stream>>>(tn, S, pd);
        _Float16* t0 = tp; tp = tc; tc = tn; tn = t0;
    }

    // -------- MLP head ------------------------------------------------------
    linear<<<cdiv(BB * 100, T), T, 0, stream>>>(S,  W1, b1, H1, 180, 100, 1);
    linear<<<cdiv(BB * 80,  T), T, 0, stream>>>(H1, W2, b2, H2, 100, 80, 1);
    linear<<<cdiv(BB * 60,  T), T, 0, stream>>>(H2, W3, b3, H3, 80, 60, 1);
    linear<<<cdiv(BB * 3,   T), T, 0, stream>>>(H3, W4, b4, (float*)d_out, 60, 3, 0);
}
